// RelationshipVisionEncoder_13082470383714
// MI455X (gfx1250) — compile-verified, matrix operand order checked by reading, not run on hardware
//
#include <hip/hip_runtime.h>
#include <math.h>

// ---------------------------------------------------------------------------
// Fused RelationshipVisionEncoder for MI455X (gfx1250, wave32, WMMA).
// One workgroup = 4 batches (40 tokens, padded to 48 rows = 3 WMMA M-tiles).
// All activations live in LDS as fp16; dense GEMMs use v_wmma_f32_16x16x32_f16
// with fp32 accumulation and 4-way N-tile blocking (one A fragment reused by
// 4 WMMAs -> 4x less LDS traffic). Attention (N=10) is VALU.
// Weights are converted fp32->fp16 once into d_ws (~3MB, stays L2-resident).
// ---------------------------------------------------------------------------

typedef _Float16 half_t;
typedef __attribute__((ext_vector_type(16))) _Float16 v16h;
typedef __attribute__((ext_vector_type(8)))  float    v8f;

#define Bt   16384
#define Nn   10
#define KNN  5
#define Dd   512
#define Hh   2
#define HDm  256
#define TBb  4            // batches per workgroup
#define RR   (TBb*Nn)     // 40 live rows
#define RT   48           // padded rows (3 tiles of 16)
#define MT   3
#define LDA  520          // LDS row stride in halves (16B-aligned, staggers banks)
#define IMGF 512.0f

// weight offsets (halves) inside d_ws
#define OFF_WQ 0
#define OFF_WK (512*512)
#define OFF_WV (1024*512)
#define OFF_WO (1536*512)
#define OFF_W1 (2048*512)
#define OFF_W2 (2560*512)
#define W_TOTAL (3072*512)   // 1,572,864 halves

__device__ __forceinline__ v16h ld16(const half_t* p0, const half_t* p1) {
  union { v16h h; float4 f[2]; } u;
  u.f[0] = *(const float4*)p0;
  u.f[1] = *(const float4*)p1;
  return u.h;
}

// ---------------------------------------------------------------------------
// Cooperative GEMM: Dst[48,512] = epilogue( A[48,512] @ W^T + bias ), fp16 out.
// W row-major [512 out][512 in] fp16 in global (L2-resident).
// EPI: 0 = bias, 1 = bias + exact GELU, 2 = bias + residual (Res fp16 LDS).
// A-fragment layout per ISA (16-bit A 16x32): lanes 0-15 hold K=0..7,16..23 of
// row M=lane; lanes 16-31 hold K=8..15,24..31. B assumed mirrored over lanes.
// C layout: VGPR v -> row v (lanes 0-15) / v+8 (lanes 16-31), col = lane&15.
// Each wave owns a group of 4 N-tiles: 1 A fragment feeds 4 WMMAs per K-step.
// ---------------------------------------------------------------------------
template<int EPI>
__device__ void gemm_tile(const half_t* __restrict__ A,
                          const half_t* __restrict__ W,
                          const float*  __restrict__ bias,
                          half_t* __restrict__ Dst,
                          const half_t* __restrict__ Res,
                          int tid) {
  const int wave = tid >> 5, lane = tid & 31;
  const int ml   = (lane & 16) ? 8 : 0;   // C row offset for hi half-wave
  const int koff = (lane & 16) ? 8 : 0;   // K offset for hi half-wave
  const int nl   = lane & 15;
  const int NG   = Dd / 64;               // 8 groups of 4 N-tiles
  for (int g = wave; g < MT * NG; g += 8) {
    const int mt = g / NG, ng = g % NG;
    const int mb = mt * 16, nb = ng * 64;
    const half_t* arow = A + (mb + nl) * LDA + koff;
    const half_t* wrow = W + (nb + nl) * Dd + koff;   // +t*16*Dd for tile t
    v8f c0 = {}, c1 = {}, c2 = {}, c3 = {};
#pragma unroll 2
    for (int kb = 0; kb < Dd; kb += 32) {
      if (kb + 32 < Dd) __builtin_prefetch(wrow + kb + 32, 0, 3); // global_prefetch_b8
      const v16h a  = ld16(arow + kb,              arow + kb + 16);
      const v16h b0 = ld16(wrow + kb,              wrow + kb + 16);
      const v16h b1 = ld16(wrow + 16 * Dd + kb,    wrow + 16 * Dd + kb + 16);
      const v16h b2 = ld16(wrow + 32 * Dd + kb,    wrow + 32 * Dd + kb + 16);
      const v16h b3 = ld16(wrow + 48 * Dd + kb,    wrow + 48 * Dd + kb + 16);
      c0 = __builtin_amdgcn_wmma_f32_16x16x32_f16(false, a, false, b0, (short)0, c0, false, false);
      c1 = __builtin_amdgcn_wmma_f32_16x16x32_f16(false, a, false, b1, (short)0, c1, false, false);
      c2 = __builtin_amdgcn_wmma_f32_16x16x32_f16(false, a, false, b2, (short)0, c2, false, false);
      c3 = __builtin_amdgcn_wmma_f32_16x16x32_f16(false, a, false, b3, (short)0, c3, false, false);
    }
#pragma unroll
    for (int t = 0; t < 4; ++t) {
      const v8f&  c  = (t == 0) ? c0 : (t == 1) ? c1 : (t == 2) ? c2 : c3;
      const int   n  = nb + t * 16 + nl;
      const float bi = bias[n];
#pragma unroll
      for (int v = 0; v < 8; ++v) {
        const int r = mb + v + ml;
        float x = c[v] + bi;
        if (EPI == 1) x = 0.5f * x * (1.0f + erff(x * 0.70710678f));  // exact GELU
        if (EPI == 2) x += (float)Res[r * LDA + n];
        Dst[r * LDA + n] = (half_t)x;
      }
    }
  }
}

// LayerNorm over D=512, one wave per row, wave32 shuffle reduction.
// dstL!=nullptr -> write fp16 LDS; else write fp32 global at dstG (stride 512).
__device__ void ln_rows(const half_t* __restrict__ src,
                        const float* __restrict__ w, const float* __restrict__ b,
                        half_t* __restrict__ dstL, float* __restrict__ dstG,
                        int tid) {
  const int wave = tid >> 5, lane = tid & 31;
  for (int r = wave; r < RR; r += 8) {
    const half_t* row = src + r * LDA;
    float s = 0.f, q = 0.f;
#pragma unroll
    for (int i = 0; i < 16; ++i) {
      const float v = (float)row[lane * 16 + i];
      s += v; q += v * v;
    }
    for (int off = 16; off; off >>= 1) {
      s += __shfl_xor(s, off, 32);
      q += __shfl_xor(q, off, 32);
    }
    const float mu  = s * (1.f / 512.f);
    const float var = q * (1.f / 512.f) - mu * mu;
    const float rs  = rsqrtf(var + 1e-5f);
    if (dstL) {
#pragma unroll
      for (int i = 0; i < 16; ++i) {
        const int d = lane * 16 + i;
        dstL[r * LDA + d] = (half_t)(((float)row[d] - mu) * rs * w[d] + b[d]);
      }
    } else {
      float4* p = (float4*)(dstG + (size_t)r * Dd + lane * 16);
#pragma unroll
      for (int ck = 0; ck < 4; ++ck) {
        float4 o;
        const int d = lane * 16 + ck * 4;
        o.x = ((float)row[d + 0] - mu) * rs * w[d + 0] + b[d + 0];
        o.y = ((float)row[d + 1] - mu) * rs * w[d + 1] + b[d + 1];
        o.z = ((float)row[d + 2] - mu) * rs * w[d + 2] + b[d + 2];
        o.w = ((float)row[d + 3] - mu) * rs * w[d + 3] + b[d + 3];
        p[ck] = o;  // global_store_b128
      }
    }
  }
}

// fp32 -> fp16 weight conversion into d_ws: [wqkv | wo | w1 | w2]
__global__ void cvt_weights(const float* __restrict__ wqkv,
                            const float* __restrict__ wo,
                            const float* __restrict__ w1,
                            const float* __restrict__ w2,
                            half_t* __restrict__ dst) {
  const int i = blockIdx.x * 256 + threadIdx.x;
  if (i >= W_TOTAL) return;
  float v;
  if      (i < OFF_WO) v = wqkv[i];
  else if (i < OFF_W1) v = wo[i - OFF_WO];
  else if (i < OFF_W2) v = w1[i - OFF_W1];
  else                 v = w2[i - OFF_W2];
  dst[i] = (half_t)v;
}

__global__ __launch_bounds__(256)
void encoder_kernel(const float* __restrict__ boxes,
                    const float* __restrict__ w_in, const float* __restrict__ b_in,
                    const float* __restrict__ bqkv, const float* __restrict__ bo,
                    const float* __restrict__ ln1w, const float* __restrict__ ln1b,
                    const float* __restrict__ b1,   const float* __restrict__ b2,
                    const float* __restrict__ ln2w, const float* __restrict__ ln2b,
                    const float* __restrict__ lnfw, const float* __restrict__ lnfb,
                    const half_t* __restrict__ hw,
                    float* __restrict__ out) {
  __shared__ __align__(16) half_t xh[RT * LDA];   // activation / residual
  __shared__ __align__(16) half_t t0[RT * LDA];   // Q -> V -> post-attn sum -> FFN hidden
  __shared__ __align__(16) half_t t1[RT * LDA];   // K -> attn out -> FFN sum
  __shared__ float att[TBb * Hh * Nn * Nn];
  __shared__ float rel[RR * 20];
  __shared__ float feats[RR * 4];

  const int tid = threadIdx.x;
  const int b0  = blockIdx.x * TBb;

  // ---- Stage A: box features -------------------------------------------
  for (int t = tid; t < RR; t += 256) {
    const int bg = b0 + t / Nn, n = t % Nn;
    const float* bx = boxes + ((size_t)bg * Nn + n) * 4;
    const float x0 = bx[0], y0 = bx[1], x1 = bx[2], y1 = bx[3];
    feats[t * 4 + 0] = (x0 + x1) * 0.5f / IMGF;
    feats[t * 4 + 1] = (y0 + y1) * 0.5f / IMGF;
    feats[t * 4 + 2] = fabsf(x1 - x0) / IMGF;
    feats[t * 4 + 3] = fabsf(y1 - y0) / IMGF;
  }
  __syncthreads();

  // ---- 5-NN (ascending distance, ties -> lower index) + rel features ----
  for (int t = tid; t < RR; t += 256) {
    const int bl = t / Nn, n = t % Nn, base = bl * Nn;
    const float cx = feats[t * 4], cy = feats[t * 4 + 1];
    float d[Nn];
    for (int j = 0; j < Nn; ++j) {
      const float dx = cx - feats[(base + j) * 4];
      const float dy = cy - feats[(base + j) * 4 + 1];
      d[j] = (j == n) ? __builtin_inff() : (dx * dx + dy * dy);
    }
    bool used[Nn];
    for (int j = 0; j < Nn; ++j) used[j] = false;
    for (int k = 0; k < KNN; ++k) {
      int bi = -1; float bd = __builtin_inff();
      for (int j = 0; j < Nn; ++j)
        if (j != n && !used[j] && d[j] < bd) { bd = d[j]; bi = j; }
      if (bi < 0) bi = (n + 1) % Nn;  // defensive (cannot happen: 9 finite >= 5)
      used[bi] = true;
      for (int c = 0; c < 4; ++c)
        rel[t * 20 + k * 4 + c] = feats[(base + bi) * 4 + c] - feats[t * 4 + c];
    }
  }
  __syncthreads();

  // ---- Embed: x = rel @ w_in^T + b_in; w_in row cached in registers -----
  for (int d = tid; d < Dd; d += 256) {
    float wr[20];
#pragma unroll
    for (int j = 0; j < 20; ++j) wr[j] = w_in[d * 20 + j];
    const float bv = b_in[d];
    for (int r = 0; r < RT; ++r) {
      float acc = 0.f;
      if (r < RR) {
        acc = bv;
#pragma unroll
        for (int j = 0; j < 20; ++j) acc += rel[r * 20 + j] * wr[j];
      }
      xh[r * LDA + d] = (half_t)acc;   // pad rows zeroed
    }
  }
  __syncthreads();

  // ---- Q, K GEMMs (WMMA) ------------------------------------------------
  gemm_tile<0>(xh, hw + OFF_WQ, bqkv,        t0, nullptr, tid);
  gemm_tile<0>(xh, hw + OFF_WK, bqkv + 512,  t1, nullptr, tid);
  __syncthreads();

  // ---- scores = Q K^T / 16 (N=10: VALU, 16B vector loads) ---------------
  for (int e = tid; e < TBb * Hh * Nn * Nn; e += 256) {
    const int j = e % Nn, i = (e / Nn) % Nn, h = (e / (Nn * Nn)) % Hh, bl = e / (Nn * Nn * Hh);
    const float4* q4 = (const float4*)(t0 + (bl * Nn + i) * LDA + h * HDm);
    const float4* k4 = (const float4*)(t1 + (bl * Nn + j) * LDA + h * HDm);
    float s = 0.f;
    for (int c = 0; c < HDm / 8; ++c) {
      union { float4 f; half_t h[8]; } uq, uk;
      uq.f = q4[c]; uk.f = k4[c];
#pragma unroll
      for (int u = 0; u < 8; ++u) s += (float)uq.h[u] * (float)uk.h[u];
    }
    att[e] = s * 0.0625f;   // 1/sqrt(256)
  }
  __syncthreads();

  // ---- softmax rows of 10 -----------------------------------------------
  for (int row = tid; row < TBb * Hh * Nn; row += 256) {
    float* a = att + row * Nn;
    float m = a[0];
    for (int j = 1; j < Nn; ++j) m = fmaxf(m, a[j]);
    float s = 0.f;
    for (int j = 0; j < Nn; ++j) { const float ev = __expf(a[j] - m); a[j] = ev; s += ev; }
    const float inv = 1.f / s;
    for (int j = 0; j < Nn; ++j) a[j] *= inv;
  }
  // ---- V GEMM (reads xh, overwrites t0; Q fully consumed pre-barrier) ---
  gemm_tile<0>(xh, hw + OFF_WV, bqkv + 1024, t0, nullptr, tid);
  __syncthreads();

  // ---- O = att @ V -> t1 (10 MACs/out, VALU) ----------------------------
  for (int e = tid; e < RR * Dd; e += 256) {
    const int r = e >> 9, d = e & 511;
    const int bl = r / Nn, n = r % Nn, h = d >> 8;
    const float* a = att + ((bl * Hh + h) * Nn + n) * Nn;
    float acc = 0.f;
    for (int j = 0; j < Nn; ++j) acc += a[j] * (float)t0[(bl * Nn + j) * LDA + d];
    t1[r * LDA + d] = (half_t)acc;
  }
  __syncthreads();

  // ---- out-proj + residual, LN1 -----------------------------------------
  gemm_tile<2>(t1, hw + OFF_WO, bo, t0, xh, tid);
  __syncthreads();
  ln_rows(t0, ln1w, ln1b, xh, nullptr, tid);
  __syncthreads();

  // ---- FFN: GELU(x W1^T + b1) W2^T + b2 + residual, LN2 -----------------
  gemm_tile<1>(xh, hw + OFF_W1, b1, t0, nullptr, tid);
  __syncthreads();
  gemm_tile<2>(t0, hw + OFF_W2, b2, t1, xh, tid);
  __syncthreads();
  ln_rows(t1, ln2w, ln2b, xh, nullptr, tid);
  __syncthreads();

  // ---- final LN -> global fp32 (float4 stores) --------------------------
  ln_rows(xh, lnfw, lnfb, nullptr, out + (size_t)b0 * Nn * Dd, tid);
}

extern "C" void kernel_launch(void* const* d_in, const int* in_sizes, int n_in,
                              void* d_out, int out_size, void* d_ws, size_t ws_size,
                              hipStream_t stream) {
  (void)in_sizes; (void)n_in; (void)out_size; (void)ws_size; // needs >= 3,145,728 B ws
  const float* boxes = (const float*)d_in[0];
  const float* w_in  = (const float*)d_in[1];
  const float* b_in  = (const float*)d_in[2];
  const float* wqkv  = (const float*)d_in[3];
  const float* bqkv  = (const float*)d_in[4];
  const float* wo    = (const float*)d_in[5];
  const float* bo    = (const float*)d_in[6];
  const float* ln1w  = (const float*)d_in[7];
  const float* ln1b  = (const float*)d_in[8];
  const float* w1    = (const float*)d_in[9];
  const float* b1    = (const float*)d_in[10];
  const float* w2    = (const float*)d_in[11];
  const float* b2    = (const float*)d_in[12];
  const float* ln2w  = (const float*)d_in[13];
  const float* ln2b  = (const float*)d_in[14];
  const float* lnfw  = (const float*)d_in[15];
  const float* lnfb  = (const float*)d_in[16];

  half_t* hw = (half_t*)d_ws;
  cvt_weights<<<(W_TOTAL + 255) / 256, 256, 0, stream>>>(wqkv, wo, w1, w2, hw);
  encoder_kernel<<<Bt / TBb, 256, 0, stream>>>(
      boxes, w_in, b_in, bqkv, bo, ln1w, ln1b, b1, b2,
      ln2w, ln2b, lnfw, lnfb, hw, (float*)d_out);
}
